// ELiCiT_83494164234569
// MI455X (gfx1250) — matrix operand order repaired and checked by reading, not compiled, not run on hardware
//
#include <hip/hip_runtime.h>
#include <math.h>

typedef _Float16 h4   __attribute__((ext_vector_type(4)));
typedef _Float16 h8   __attribute__((ext_vector_type(8)));
typedef _Float16 v16h __attribute__((ext_vector_type(16)));
typedef float    v8f  __attribute__((ext_vector_type(8)));
typedef float    v4f  __attribute__((ext_vector_type(4)));

#define TBL_HALF 65536
#define FDIM 64
#define DDIM 128
#define WSTR 136          // padded LDS row stride in halfs (bank-conflict-free b128 loads)
#define MT   128          // rows per tile (8 waves x 16 rows)
#define GRIDMAX 1024      // persistent blocks; each processes nTiles/GRIDMAX tiles

__device__ __forceinline__ v16h cat8(h8 lo, h8 hi) {
  return __builtin_shufflevector(lo, hi, 0,1,2,3,4,5,6,7,8,9,10,11,12,13,14,15);
}

extern "C" __global__ __launch_bounds__(256)
void elicit_fused(const int* __restrict__ idxs0, const int* __restrict__ idxs1,
                  const float* __restrict__ init_feat, const float* __restrict__ values,
                  const float* __restrict__ W1, const float* __restrict__ b1,
                  const float* __restrict__ W2, const float* __restrict__ b2,
                  const float* __restrict__ W3, const float* __restrict__ b3,
                  const float* __restrict__ W4, const float* __restrict__ b4,
                  const float* __restrict__ g1, const float* __restrict__ be1,
                  const float* __restrict__ g2, const float* __restrict__ be2,
                  const float* __restrict__ g3, const float* __restrict__ be3,
                  float* __restrict__ out, int N)
{
  extern __shared__ _Float16 lds[];
  _Float16* wlds = lds;                                   // 4 * 128 * WSTR halfs (weights, f16)
  _Float16* alds = lds + 4 * DDIM * WSTR;                 // 8 waves * 16 * WSTR halfs (activations)
  float*    plds = (float*)(alds + 8 * 16 * WSTR);        // bias[4][128], g[3][128], e[3][128]

  const int tid   = threadIdx.x;
  const int wave  = tid >> 5;
  const int lane  = tid & 31;
  const int lhalf = lane >> 4;                // which 16-lane half of the wave
  const int l16   = lane & 15;

  const float* Ws[4] = {W1, W2, W3, W4};
  const float* Bv[4] = {b1, b2, b3, b4};
  const float* Gv[3] = {g1, g2, g3};
  const float* Ev[3] = {be1, be2, be3};

  // ---- stage all four weight matrices into LDS as f16 (row-major = B-fragment layout) ----
  #pragma unroll
  for (int l = 0; l < 4; ++l) {
    const float* W = Ws[l];
    _Float16* wl = wlds + l * DDIM * WSTR;
    #pragma unroll
    for (int it = 0; it < 16; ++it) {
      int i   = it * 1024 + tid * 4;          // covers 128*128 floats with 256 threads
      int row = i >> 7, col = i & 127;
      v4f w = *(const v4f*)(W + i);
      h4 hw = { (_Float16)w.x, (_Float16)w.y, (_Float16)w.z, (_Float16)w.w };
      *(h4*)(wl + row * WSTR + col) = hw;
    }
  }
  // ---- stage per-layer bias / gamma / beta vectors into LDS (read via DS pipe per tile) ----
  if (tid < 128) {
    #pragma unroll
    for (int l = 0; l < 4; ++l) plds[l * 128 + tid] = Bv[l][tid];
    #pragma unroll
    for (int l = 0; l < 3; ++l) {
      plds[512 + l * 128 + tid] = Gv[l][tid];
      plds[896 + l * 128 + tid] = Ev[l][tid];
    }
  }

  // ---- hoist values[4][f] for this lane's epilogue columns (loop-invariant) ----
  float vv[4][4];
  #pragma unroll
  for (int nt = 0; nt < 4; ++nt) {
    int f = nt * 16 + l16;
    #pragma unroll
    for (int k = 0; k < 4; ++k) vv[nt][k] = values[k * FDIM + f];
  }

  __syncthreads();   // weights + params ready; from here each wave is fully independent

  const int koff  = lhalf * 8;   // A-fragment K sub-offset per ISA 16-bit A layout
  _Float16* act   = alds + wave * 16 * WSTR;
  const int nTiles = N / MT;

  for (int tile = blockIdx.x; tile < nTiles; tile += gridDim.x) {
    const int rowBase = tile * MT + wave * 16;
    const int myrow   = rowBase + l16;

    // ---- gather this wave's 16 input rows into its private LDS act tile ----
    {
      const long gidx   = lhalf ? ((long)idxs1[myrow] + TBL_HALF) : (long)idxs0[myrow];
      const float* gsrc = init_feat + gidx * FDIM;
      _Float16* dst = act + l16 * WSTR + lhalf * FDIM;  // half 0: cols 0-63, half 1: 64-127
      #pragma unroll
      for (int j = 0; j < FDIM; j += 4) {
        v4f v = *(const v4f*)(gsrc + j);
        h4 hv = { (_Float16)v.x, (_Float16)v.y, (_Float16)v.z, (_Float16)v.w };
        *(h4*)(dst + j) = hv;
      }
    }

    // ---- prefetch next tile's gather rows behind this tile's compute ----
    {
      int nextTile = tile + gridDim.x;
      if (nextTile < nTiles) {
        int nrow = nextTile * MT + wave * 16 + l16;
        long ni  = lhalf ? ((long)idxs1[nrow] + TBL_HALF) : (long)idxs0[nrow];
        __builtin_prefetch(init_feat + ni * FDIM, 0, 0);   // global_prefetch_b8
      }
    }

    #pragma unroll
    for (int layer = 0; layer < 4; ++layer) {
      // ---- A fragments (16x32 f16 each) for the 4 K-chunks, loaded once per layer ----
      v16h afrag[4];
      const _Float16* arow = act + l16 * WSTR;
      #pragma unroll
      for (int kc = 0; kc < 4; ++kc) {
        h8 lo = *(const h8*)(arow + kc * 32 + koff);        // K = koff .. koff+7
        h8 hi = *(const h8*)(arow + kc * 32 + koff + 16);   // K = koff+16 .. koff+23
        afrag[kc] = cat8(lo, hi);
      }

      const _Float16* wl = wlds + layer * DDIM * WSTR;
      v8f acc[8];

      // ---- GEMM: 8 n-tiles x 4 k-chunks of v_wmma_f32_16x16x32_f16 ----
      #pragma unroll
      for (int nt = 0; nt < 8; ++nt) {
        float bv = plds[layer * 128 + nt * 16 + l16];       // bias depends only on column
        v8f c = { bv, bv, bv, bv, bv, bv, bv, bv };
        const _Float16* wrow = wl + (nt * 16 + l16) * WSTR + lhalf * 8;
        #pragma unroll
        for (int kc = 0; kc < 4; ++kc) {
          h8 lo = *(const h8*)(wrow + kc * 32);
          h8 hi = *(const h8*)(wrow + kc * 32 + 16);
          v16h bfrag = cat8(lo, hi);                        // B = W^T chunk (K x 16)
          c = __builtin_amdgcn_wmma_f32_16x16x32_f16(false, afrag[kc], false, bfrag,
                                                     (short)0, c, false, false);
        }
        acc[nt] = c;
      }

      if (layer < 3) {
        // ---- register-resident LayerNorm + exact GELU ----
        float s[8], q[8];
        #pragma unroll
        for (int r = 0; r < 8; ++r) { s[r] = 0.f; q[r] = 0.f; }
        #pragma unroll
        for (int nt = 0; nt < 8; ++nt)
          #pragma unroll
          for (int r = 0; r < 8; ++r) {
            float x = acc[nt][r];
            s[r] += x; q[r] += x * x;
          }
        #pragma unroll
        for (int d = 1; d < 16; d <<= 1) {
          #pragma unroll
          for (int r = 0; r < 8; ++r) {
            s[r] += __shfl_xor(s[r], d, 32);
            q[r] += __shfl_xor(q[r], d, 32);
          }
        }
        #pragma unroll
        for (int nt = 0; nt < 8; ++nt) {
          int col = nt * 16 + l16;
          float gg = plds[512 + layer * 128 + col];
          float ee = plds[896 + layer * 128 + col];
          #pragma unroll
          for (int r = 0; r < 8; ++r) {
            float mean = s[r] * (1.f / 128.f);
            float var  = q[r] * (1.f / 128.f) - mean * mean;
            float x = (acc[nt][r] - mean) * rsqrtf(var + 1e-5f);
            float y = x * gg + ee;
            y = 0.5f * y * (1.f + erff(y * 0.70710678118654752f));
            act[(r + 8 * lhalf) * WSTR + col] = (_Float16)y;   // in-order same-wave LDS
          }
        }
      } else {
        // ---- epilogue: p1 = cols 0..63 (nt 0..3), p2 = cols 64..127 (nt 4..7) ----
        float part[8];
        #pragma unroll
        for (int r = 0; r < 8; ++r) part[r] = 0.f;
        #pragma unroll
        for (int nt = 0; nt < 4; ++nt) {
          int f = nt * 16 + l16;
          #pragma unroll
          for (int r = 0; r < 8; ++r) {
            float p1 = acc[nt][r], p2 = acc[nt + 4][r];
            float pa = 0.5f + p1, pb = 0.5f - p1;
            float pc = 0.5f + p2, pd = 0.5f - p2;
            float ft = vv[nt][0] * pa * pc + vv[nt][1] * pa * pd
                     + vv[nt][2] * pb * pc + vv[nt][3] * pb * pd;
            part[r] += ft;
            long n = rowBase + r + 8 * lhalf;
            out[(long)N + n * FDIM + f] = ft;                 // feats
          }
        }
        #pragma unroll
        for (int d = 1; d < 16; d <<= 1)
          #pragma unroll
          for (int r = 0; r < 8; ++r)
            part[r] += __shfl_xor(part[r], d, 32);
        if (l16 == 0) {
          #pragma unroll
          for (int r = 0; r < 8; ++r)
            out[rowBase + r + 8 * lhalf] = part[r];           // vals
        }
      }
    }
  }
}

extern "C" void kernel_launch(void* const* d_in, const int* in_sizes, int n_in,
                              void* d_out, int out_size, void* d_ws, size_t ws_size,
                              hipStream_t stream) {
  const int*   idxs0     = (const int*)d_in[0];
  const int*   idxs1     = (const int*)d_in[1];
  const float* init_feat = (const float*)d_in[2];
  const float* values    = (const float*)d_in[3];
  const float* W1 = (const float*)d_in[4];   const float* b1 = (const float*)d_in[5];
  const float* W2 = (const float*)d_in[6];   const float* b2 = (const float*)d_in[7];
  const float* W3 = (const float*)d_in[8];   const float* b3 = (const float*)d_in[9];
  const float* W4 = (const float*)d_in[10];  const float* b4 = (const float*)d_in[11];
  const float* g1 = (const float*)d_in[12];  const float* be1 = (const float*)d_in[13];
  const float* g2 = (const float*)d_in[14];  const float* be2 = (const float*)d_in[15];
  const float* g3 = (const float*)d_in[16];  const float* be3 = (const float*)d_in[17];
  float* out = (float*)d_out;

  int N = in_sizes[0];
  int nTiles = N / MT;
  int grid = nTiles < GRIDMAX ? nTiles : GRIDMAX;

  size_t ldsBytes = (size_t)(4 * DDIM * WSTR + 8 * 16 * WSTR) * sizeof(_Float16)
                  + (size_t)1280 * sizeof(float);   // ~175 KB of the 320 KB WGP LDS

  hipFuncSetAttribute((const void*)elicit_fused,
                      hipFuncAttributeMaxDynamicSharedMemorySize, (int)ldsBytes);

  elicit_fused<<<grid, 256, ldsBytes, stream>>>(
      idxs0, idxs1, init_feat, values,
      W1, b1, W2, b2, W3, b3, W4, b4,
      g1, be1, g2, be2, g3, be3,
      out, N);
}
